// UV_Aggregator_70102456205704
// MI455X (gfx1250) — matrix operand order repaired and checked by reading, not compiled, not run on hardware
//
#include <hip/hip_runtime.h>
#include <hip/hip_bf16.h>

typedef _Float16 f16;
typedef __attribute__((ext_vector_type(16))) _Float16 v16h;
typedef __attribute__((ext_vector_type(8)))  _Float16 v8h;
typedef __attribute__((ext_vector_type(8)))  float    v8f;

#define B_SZ 16384
#define L_SZ 50

// LDS strides (in halves), padded to avoid 64-bank conflicts
#define XS  136   // rows of width 128
#define WS2 72    // rows of width 64

// LDS half-offsets (all multiples of 8 -> 16B aligned)
#define OFF_X   0        // 64*136 = 8704 halves (X, later reused for H2)
#define OFF_W1  8704     // 64*136 (w1T / a1T)
#define OFF_W2  17408    // 64*72  (w2T / a2T)
#define OFF_O1  22016    // 64*72  (O1, later H1)
#define OFF_O2  26624    // 64*72  (O2, persists for aggregation)
#define OFF_U   31232    // 64 halves user embedding
#define SMEM_H  31296    // total halves = 62592 bytes

// ---- WMMA fragment loaders (CDNA5 16x16x32 f16 layouts, ISA 7.12.2) ----
// A (16x32, MxK): lane = h*16 + m ; VGPR0-3 hold K = h*8 + 0..7, VGPR4-7 hold K = 16 + h*8 + 0..7
__device__ __forceinline__ v16h load_a_frag(const f16* buf, int stride, int row, int kofs, int h) {
  const f16* p = buf + row * stride + kofs + h * 8;
  v8h lo = *(const v8h*)(p);
  v8h hi = *(const v8h*)(p + 16);
  return __builtin_shufflevector(lo, hi, 0,1,2,3,4,5,6,7,8,9,10,11,12,13,14,15);
}
// B (32x16, KxN): lane = h*16 + n ; elements 0-7 hold K = h*8 + 0..7, elements 8-15 hold K = 16 + h*8 + 0..7  (weights stored transposed [n][k])
__device__ __forceinline__ v16h load_b_frag(const f16* buf, int stride, int n, int kofs, int h) {
  const f16* p = buf + n * stride + kofs + h * 8;
  v8h lo = *(const v8h*)(p);
  v8h hi = *(const v8h*)(p + 16);
  return __builtin_shufflevector(lo, hi, 0,1,2,3,4,5,6,7,8,9,10,11,12,13,14,15);
}

// One dense layer for a 16-row tile: out[16x64] = relu(A[16xK] * W[Kx64] + bias)
// If uBuf != nullptr, K-slices >= 64 come from the row-broadcast user embedding (concat).
__device__ __forceinline__ void gemm_layer(
    const f16* aBuf, int astride, const f16* uBuf,
    const f16* wBuf, int wstride, int nk /* K/32 */,
    const float* bias, f16* outBuf, int ostride,
    int rowbase, int lane)
{
  const int n0 = lane & 15;
  const int h  = lane >> 4;
  v8f acc[4];
#pragma unroll
  for (int nt = 0; nt < 4; ++nt) {
    float bv = bias[nt * 16 + n0];
    v8f tmp = {bv, bv, bv, bv, bv, bv, bv, bv};
    acc[nt] = tmp;
  }
#pragma unroll
  for (int ks = 0; ks < nk; ++ks) {
    v16h a;
    if (uBuf != nullptr && ks >= 2)
      a = load_a_frag(uBuf, 0, 0, (ks - 2) * 32, h);       // broadcast rows (u_b)
    else
      a = load_a_frag(aBuf, astride, rowbase + n0, ks * 32, h);
#pragma unroll
    for (int nt = 0; nt < 4; ++nt) {
      v16h b = load_b_frag(wBuf, wstride, nt * 16 + n0, ks * 32, h);
      acc[nt] = __builtin_amdgcn_wmma_f32_16x16x32_f16(
          false, a, false, b, (short)0, acc[nt], false, false);
    }
  }
  // D layout: element r -> row = rowbase + r + h*8, col = nt*16 + n0 ; relu + cvt to f16
#pragma unroll
  for (int nt = 0; nt < 4; ++nt) {
#pragma unroll
    for (int r = 0; r < 8; ++r) {
      float v = acc[nt][r];
      v = v > 0.f ? v : 0.f;
      outBuf[(rowbase + r + h * 8) * ostride + nt * 16 + n0] = (f16)v;
    }
  }
}

// Pre-kernel: fp32 [K][64] -> f16 transposed [64][K] (contiguous writes)
__global__ void cvt_transpose_kernel(const float* __restrict__ src, f16* __restrict__ dst, int K) {
  int t = blockIdx.x * blockDim.x + threadIdx.x;
  if (t < 64 * K) {
    int n = t / K, k = t - n * K;
    dst[t] = (f16)src[k * 64 + n];
  }
}

__global__ __launch_bounds__(128) void uv_agg_kernel(
    const int* __restrict__ nodes, const int* __restrict__ hist_uv, const int* __restrict__ hist_r,
    const float* __restrict__ v2e, const float* __restrict__ u2e, const float* __restrict__ r2e,
    const f16* __restrict__ wT,
    const float* __restrict__ b1, const float* __restrict__ b2,
    const float* __restrict__ ab1, const float* __restrict__ ab2,
    const float* __restrict__ a3, const float* __restrict__ ab3,
    float* __restrict__ out)
{
  __shared__ f16   sh[SMEM_H];
  __shared__ float sf[128];   // [0,64) logits ; [64,128) attention weights

  const int t    = threadIdx.x;
  const int lane = t & 31;
  const int wid  = t >> 5;
  const int b    = blockIdx.x;

  f16* sX  = sh + OFF_X;
  f16* sW1 = sh + OFF_W1;
  f16* sW2 = sh + OFF_W2;
  f16* sO1 = sh + OFF_O1;   // also H1
  f16* sO2 = sh + OFF_O2;
  f16* sU  = sh + OFF_U;
  f16* sH2 = sh + OFF_X;    // reuse X region with stride WS2

  // ---- stage user embedding ----
  if (t < 64) sU[t] = (f16)u2e[(size_t)nodes[b] * 64 + t];

  // ---- gather item + rating embeddings -> sX [64][XS] (concat along K) ----
  for (int l = wid; l < L_SZ; l += 4) {
    int iv = hist_uv[b * L_SZ + l];
    int ir = hist_r[b * L_SZ + l];
    ir = ir > 0 ? ir - 1 : (ir < 0 ? ir + 1 : ir);
    const float* pv = v2e + (size_t)iv * 64 + lane * 2;
    const float* pr = r2e + (size_t)ir * 64 + lane * 2;
    f16* px = sX + l * XS;
    px[lane * 2]          = (f16)pv[0];
    px[lane * 2 + 1]      = (f16)pv[1];
    px[64 + lane * 2]     = (f16)pr[0];
    px[64 + lane * 2 + 1] = (f16)pr[1];
  }
  for (int i = t; i < (64 - L_SZ) * XS; i += 128) sX[L_SZ * XS + i] = (f16)0.f;

  // ---- stage weights phase 1: w1T, w2T (f16, 4B at a time) ----
  {
    const unsigned* g1 = (const unsigned*)(wT);           // w1T [64][128]
    for (int i = t; i < 64 * 128 / 2; i += 128) {
      int e = i * 2, n = e >> 7, k = e & 127;
      *(unsigned*)(sW1 + n * XS + k) = g1[i];
    }
    const unsigned* g2 = (const unsigned*)(wT + 8192);    // w2T [64][64]
    for (int i = t; i < 64 * 64 / 2; i += 128) {
      int e = i * 2, n = e >> 6, k = e & 63;
      *(unsigned*)(sW2 + n * WS2 + k) = g2[i];
    }
  }
  __syncthreads();

  const int rowbase = wid * 16;
  // L1: x = relu([e_uv,e_r] @ w1 + b1)
  gemm_layer(sX, XS, nullptr, sW1, XS, 4, b1, sO1, WS2, rowbase, lane);
  __syncthreads();
  // L2: o = relu(x @ w2 + b2)
  gemm_layer(sO1, WS2, nullptr, sW2, WS2, 2, b2, sO2, WS2, rowbase, lane);
  __syncthreads();

  // ---- restage weights phase 2: a1T, a2T ----
  {
    const unsigned* g1 = (const unsigned*)(wT + 12288);   // a1T [64][128]
    for (int i = t; i < 64 * 128 / 2; i += 128) {
      int e = i * 2, n = e >> 7, k = e & 127;
      *(unsigned*)(sW1 + n * XS + k) = g1[i];
    }
    const unsigned* g2 = (const unsigned*)(wT + 20480);   // a2T [64][64]
    for (int i = t; i < 64 * 64 / 2; i += 128) {
      int e = i * 2, n = e >> 6, k = e & 63;
      *(unsigned*)(sW2 + n * WS2 + k) = g2[i];
    }
  }
  __syncthreads();

  // L3: h = relu([o, u_b] @ a1 + ab1)   (K-slices 2..3 broadcast from sU)
  gemm_layer(sO2, WS2, sU, sW1, XS, 4, ab1, sO1, WS2, rowbase, lane);
  __syncthreads();
  // L4: h = relu(h @ a2 + ab2) -> sH2 (reuses X region)
  gemm_layer(sO1, WS2, nullptr, sW2, WS2, 2, ab2, sH2, WS2, rowbase, lane);
  __syncthreads();

  // ---- logits: h @ a3 + ab3 ----
  if (t < L_SZ) {
    float s = ab3[0];
    const f16* ph = sH2 + t * WS2;
    for (int k = 0; k < 64; ++k) s += (float)ph[k] * a3[k];
    sf[t] = s;
  }
  __syncthreads();
  // ---- softmax over L (tiny, serial) ----
  if (t == 0) {
    float m = sf[0];
    for (int l = 1; l < L_SZ; ++l) m = fmaxf(m, sf[l]);
    float sum = 0.f;
    for (int l = 0; l < L_SZ; ++l) { float e = __expf(sf[l] - m); sf[64 + l] = e; sum += e; }
    float inv = 1.f / sum;
    for (int l = 0; l < L_SZ; ++l) sf[64 + l] *= inv;
  }
  __syncthreads();
  // ---- attention-weighted aggregation of o ----
  if (t < 64) {
    float acc = 0.f;
    for (int l = 0; l < L_SZ; ++l) acc += sf[64 + l] * (float)sO2[l * WS2 + t];
    out[(size_t)b * 64 + t] = acc;
  }
}

extern "C" void kernel_launch(void* const* d_in, const int* in_sizes, int n_in,
                              void* d_out, int out_size, void* d_ws, size_t ws_size,
                              hipStream_t stream) {
  const int*   nodes = (const int*)d_in[0];
  const int*   huv   = (const int*)d_in[1];
  const int*   hr    = (const int*)d_in[2];
  const float* v2e   = (const float*)d_in[3];
  const float* u2e   = (const float*)d_in[4];
  const float* r2e   = (const float*)d_in[5];
  const float* w1    = (const float*)d_in[6];
  const float* b1    = (const float*)d_in[7];
  const float* w2    = (const float*)d_in[8];
  const float* b2    = (const float*)d_in[9];
  const float* a1    = (const float*)d_in[10];
  const float* ab1   = (const float*)d_in[11];
  const float* a2    = (const float*)d_in[12];
  const float* ab2   = (const float*)d_in[13];
  const float* a3    = (const float*)d_in[14];
  const float* ab3   = (const float*)d_in[15];

  f16* wT = (f16*)d_ws;  // 24576 halves = 48KB of scratch used
  cvt_transpose_kernel<<<(64 * 128 + 255) / 256, 256, 0, stream>>>(w1, wT,          128);
  cvt_transpose_kernel<<<(64 *  64 + 255) / 256, 256, 0, stream>>>(w2, wT + 8192,    64);
  cvt_transpose_kernel<<<(64 * 128 + 255) / 256, 256, 0, stream>>>(a1, wT + 12288,  128);
  cvt_transpose_kernel<<<(64 *  64 + 255) / 256, 256, 0, stream>>>(a2, wT + 20480,   64);

  uv_agg_kernel<<<B_SZ, 128, 0, stream>>>(nodes, huv, hr, v2e, u2e, r2e, wT,
                                          b1, b2, ab1, ab2, a3, ab3, (float*)d_out);
}
